// MrBertForSequenceClassification_28819230556526
// MI455X (gfx1250) — compile-verified, matrix operand order checked by reading, not run on hardware
//
#include <hip/hip_runtime.h>
#include <hip/hip_bf16.h>

typedef __attribute__((ext_vector_type(16))) __bf16 v16bf;
typedef __attribute__((ext_vector_type(8)))  float  v8f;

namespace {
constexpr int Bc = 4, Sc = 512, Hc = 768, Lc = 12, NHc = 12, DHc = 64, FFc = 3072;
constexpr int Mc = Bc * Sc;                 // 2048 rows
constexpr float EPSc = 1e-12f;
constexpr float GATE_Kc = -10.0f;
constexpr int GATE_LAYERc = 5;
constexpr int GTM = 64, GTN = 128, GTK = 32, LSTR = 40; // LDS row stride (halves) -> 80B, 16B aligned
}

// ---- f32 -> bf16 conversion helpers (packed V_CVT_PK_BF16_F32 when available)
__device__ __forceinline__ unsigned short f2bf_manual(float f) {
  unsigned int u = __builtin_bit_cast(unsigned int, f);
  unsigned int r = u + 0x7fffu + ((u >> 16) & 1u);
  return (unsigned short)(r >> 16);
}
__device__ __forceinline__ unsigned int f2bf_pk(float a, float b) {
#if __has_builtin(__builtin_amdgcn_cvt_pk_bf16_f32)
  auto r = __builtin_amdgcn_cvt_pk_bf16_f32(a, b);
  return __builtin_bit_cast(unsigned int, r);
#else
  return (unsigned int)f2bf_manual(a) | ((unsigned int)f2bf_manual(b) << 16);
#endif
}
__device__ __forceinline__ unsigned short f2bf_s(float a) {
  return (unsigned short)(f2bf_pk(a, a) & 0xffffu);
}

union Frag {
  unsigned short u16[16];
  unsigned int u32[8];
  uint4 q[2];
  v16bf v;
};

__device__ __forceinline__ v8f vzero8() {
  v8f z = {0.f, 0.f, 0.f, 0.f, 0.f, 0.f, 0.f, 0.f};
  return z;
}

__device__ __forceinline__ v8f wmma_bf16(const Frag& a, const Frag& b, v8f c) {
  return __builtin_amdgcn_wmma_f32_16x16x32_bf16(false, a.v, false, b.v,
                                                 (short)0, c, false, false);
}

__device__ __forceinline__ unsigned lds_off(const void* p) {
  return (unsigned)(unsigned long long)p;   // LDS aperture: low 32 bits = DS offset
}

// ---------------------------------------------------------------------------
// Weight convert + transpose: W[K,N] f32 -> Wt[N,K] bf16 (32x32 LDS tiles)
// ---------------------------------------------------------------------------
__global__ __launch_bounds__(256) void convert_transpose_w(
    const float* __restrict__ W, unsigned short* __restrict__ Wt, int K, int N)
{
  __shared__ float tile[32][33];
  const int n0 = blockIdx.x * 32, k0 = blockIdx.y * 32;
  const int tx = threadIdx.x & 31, ty = threadIdx.x >> 5;  // 32 x 8
  for (int r = ty; r < 32; r += 8)
    tile[r][tx] = W[(size_t)(k0 + r) * N + n0 + tx];
  __syncthreads();
  for (int r = ty; r < 32; r += 8)
    Wt[(size_t)(n0 + r) * K + k0 + tx] = f2bf_s(tile[tx][r]);
}

// ---------------------------------------------------------------------------
// GEMM: C[M,N] = A[M,K](bf16) @ Wt[N,K](bf16)^T + bias (+ epilogue)
//   EPI 0: bias            -> bf16 out
//   EPI 1: bias + GELU     -> bf16 out
//   EPI 2: bias + residual -> f32 out
// 256 threads (8 waves 2x4), 64x128 tile, K step 32.
// Staging: global_load_async_to_lds_b128, double-buffered, s_wait_asynccnt.
// ---------------------------------------------------------------------------
template <int EPI>
__global__ __launch_bounds__(256) void gemm_bf16_wmma(
    const unsigned short* __restrict__ A,
    const unsigned short* __restrict__ Wt,
    const float* __restrict__ bias, const float* __restrict__ R,
    void* __restrict__ Cout, int N, int K)
{
  __shared__ __align__(16) unsigned short lA[2][GTM][LSTR];
  __shared__ __align__(16) unsigned short lB[2][GTN][LSTR];

  const int t = threadIdx.x;
  const int lane = t & 31;
  const int wave = t >> 5;
  const int wm = wave >> 2;   // 0..1
  const int wn = wave & 3;    // 0..3
  const int m0 = blockIdx.x * GTM;
  const int n0 = blockIdx.y * GTN;
  const int g = lane >> 4;
  const int r16 = lane & 15;

  // async staging map: one b128 (16B = 8 bf16) per op
  const int aRow = t >> 2, aCh = t & 3;        // A: 64 rows x 4 chunks
  const unsigned short* aG = A + (size_t)(m0 + aRow) * K + aCh * 8;

  auto issue_tile = [&](int buf, int k0) {
    asm volatile("global_load_async_to_lds_b128 %0, %1, off"
                 :: "v"(lds_off(&lA[buf][aRow][aCh * 8])), "v"(aG + k0)
                 : "memory");
#pragma unroll
    for (int j = 0; j < 2; ++j) {
      const int idx = t + 256 * j;             // B: 128 rows x 4 chunks
      const int row = idx >> 2, ch = idx & 3;
      const unsigned short* gp = Wt + (size_t)(n0 + row) * K + k0 + ch * 8;
      asm volatile("global_load_async_to_lds_b128 %0, %1, off"
                   :: "v"(lds_off(&lB[buf][row][ch * 8])), "v"(gp)
                   : "memory");
    }
  };

  v8f acc[2][2];
  acc[0][0] = vzero8(); acc[0][1] = vzero8();
  acc[1][0] = vzero8(); acc[1][1] = vzero8();

  const int nt = K / GTK;
  issue_tile(0, 0);
  issue_tile(1, GTK);

  for (int it = 0; it < nt; ++it) {
    if (it + 1 < nt)
      asm volatile("s_wait_asynccnt 0x3" ::: "memory");  // current tile landed, next in flight
    else
      asm volatile("s_wait_asynccnt 0x0" ::: "memory");
    __syncthreads();

    const int buf = it & 1;
    Frag af[2], bf2[2];
#pragma unroll
    for (int ta = 0; ta < 2; ++ta) {
      const int mr = wm * 32 + ta * 16 + r16;
      af[ta].q[0] = *(const uint4*)&lA[buf][mr][g * 8];
      af[ta].q[1] = *(const uint4*)&lA[buf][mr][16 + g * 8];
    }
#pragma unroll
    for (int tb = 0; tb < 2; ++tb) {
      const int nc = wn * 32 + tb * 16 + r16;
      bf2[tb].q[0] = *(const uint4*)&lB[buf][nc][g * 8];
      bf2[tb].q[1] = *(const uint4*)&lB[buf][nc][16 + g * 8];
    }
#pragma unroll
    for (int ta = 0; ta < 2; ++ta)
#pragma unroll
      for (int tb = 0; tb < 2; ++tb)
        acc[ta][tb] = wmma_bf16(af[ta], bf2[tb], acc[ta][tb]);
    __syncthreads();

    if (it + 2 < nt) issue_tile(buf, (it + 2) * GTK);
  }

  float* Cf = (float*)Cout;
  unsigned short* Cb = (unsigned short*)Cout;
#pragma unroll
  for (int ta = 0; ta < 2; ++ta) {
#pragma unroll
    for (int tb = 0; tb < 2; ++tb) {
      const int col = n0 + wn * 32 + tb * 16 + r16;
      const float bcol = bias[col];
#pragma unroll
      for (int r = 0; r < 8; ++r) {
        const int row = m0 + wm * 32 + ta * 16 + g * 8 + r;
        float x = acc[ta][tb][r] + bcol;
        if (EPI == 1) x = 0.5f * x * (1.0f + erff(x * 0.70710678118654752f));
        if (EPI == 2) {
          x += R[(size_t)row * N + col];
          Cf[(size_t)row * N + col] = x;
        } else {
          Cb[(size_t)row * N + col] = f2bf_s(x);
        }
      }
    }
  }
}

// ---------------------------------------------------------------------------
// Fused attention per (batch, head, 16-query tile). q/k/v bf16 in, ctx bf16 out.
// One wave32 workgroup; 16x512 f32 scores in LDS.
// ---------------------------------------------------------------------------
__global__ __launch_bounds__(32) void attention_wmma(
    const unsigned short* __restrict__ q, const unsigned short* __restrict__ k,
    const unsigned short* __restrict__ v, const float* __restrict__ kbias,
    unsigned short* __restrict__ ctx, int use_softmax1)
{
  __shared__ __align__(16) float sc[16][Sc];  // 32 KB

  const int bh = blockIdx.x;
  const int b = bh / NHc, h = bh % NHc;
  const int q0 = blockIdx.y * 16;
  const int lane = threadIdx.x;
  const int g = lane >> 4, r16 = lane & 15;
  const float scale = 0.125f;  // 1/sqrt(64)

  // Q fragments: two contiguous 16B runs per lane, straight from global bf16
  Frag qf[2];
  {
    const unsigned short* qp = q + ((size_t)b * Sc + q0 + r16) * Hc + h * DHc;
#pragma unroll
    for (int s = 0; s < 2; ++s) {
      qf[s].q[0] = *(const uint4*)(qp + s * 32 + g * 8);
      qf[s].q[1] = *(const uint4*)(qp + s * 32 + 16 + g * 8);
    }
  }

  // scores = (Q @ K^T) * scale + kbias
  for (int kt = 0; kt < Sc / 16; ++kt) {
    v8f acc = vzero8();
    const unsigned short* kp = k + ((size_t)b * Sc + kt * 16 + r16) * Hc + h * DHc;
#pragma unroll
    for (int s = 0; s < 2; ++s) {
      Frag bf;
      bf.q[0] = *(const uint4*)(kp + s * 32 + g * 8);
      bf.q[1] = *(const uint4*)(kp + s * 32 + 16 + g * 8);
      acc = wmma_bf16(qf[s], bf, acc);
    }
    const float kbv = kbias[b * Sc + kt * 16 + r16];
#pragma unroll
    for (int r = 0; r < 8; ++r)
      sc[g * 8 + r][kt * 16 + r16] = acc[r] * scale + kbv;
  }
  __syncthreads();

  // softmax (or softmax1: e / (1 + sum e)) per row
  {
    const int row = lane & 15, half = lane >> 4;
    float* sr = &sc[row][half * (Sc / 2)];
    float mx = -3.0e38f;
    for (int j = 0; j < Sc / 2; ++j) mx = fmaxf(mx, sr[j]);
    mx = fmaxf(mx, __shfl_xor(mx, 16, 32));
    float sum = 0.f;
    for (int j = 0; j < Sc / 2; ++j) {
      float e = __expf(sr[j] - mx);
      sr[j] = e;
      sum += e;
    }
    sum += __shfl_xor(sum, 16, 32);
    const float inv = 1.f / (use_softmax1 ? (1.f + sum) : sum);
    for (int j = 0; j < Sc / 2; ++j) sr[j] *= inv;
  }
  __syncthreads();

  // ctx = probs[16x512] @ V[512x64]
  for (int dt = 0; dt < DHc / 16; ++dt) {
    v8f acc = vzero8();
    for (int st = 0; st < Sc / 32; ++st) {
      Frag af, bf;
      const float* pr = &sc[r16][st * 32];
#pragma unroll
      for (int i = 0; i < 4; ++i)
        af.u32[i] = f2bf_pk(pr[g * 8 + 2 * i], pr[g * 8 + 2 * i + 1]);
#pragma unroll
      for (int i = 0; i < 4; ++i)
        af.u32[4 + i] = f2bf_pk(pr[16 + g * 8 + 2 * i], pr[16 + g * 8 + 2 * i + 1]);
      const unsigned short* vp =
          v + ((size_t)b * Sc + st * 32) * Hc + h * DHc + dt * 16 + r16;
#pragma unroll
      for (int e = 0; e < 16; ++e) {
        const int ke = (e < 8 ? 0 : 16) + g * 8 + (e & 7);
        bf.u16[e] = vp[(size_t)ke * Hc];
      }
      acc = wmma_bf16(af, bf, acc);
    }
#pragma unroll
    for (int r = 0; r < 8; ++r)
      ctx[((size_t)b * Sc + q0 + g * 8 + r) * Hc + h * DHc + dt * 16 + r16] =
          f2bf_s(acc[r]);
  }
}

// ---------------------------------------------------------------------------
// Row LayerNorm with optional fused bf16 copy of the output.
// ---------------------------------------------------------------------------
__global__ __launch_bounds__(256) void layernorm_rows(
    const float* __restrict__ X, const float* __restrict__ sN,
    const float* __restrict__ bN, float* __restrict__ Y,
    unsigned short* __restrict__ Ybf)
{
  __shared__ float red[256];
  const int row = blockIdx.x, t = threadIdx.x;
  const float* x = X + (size_t)row * Hc;
  float s = 0.f;
  for (int i = t; i < Hc; i += 256) s += x[i];
  red[t] = s; __syncthreads();
  for (int o = 128; o > 0; o >>= 1) { if (t < o) red[t] += red[t + o]; __syncthreads(); }
  const float mean = red[0] * (1.0f / Hc);
  __syncthreads();
  float vv = 0.f;
  for (int i = t; i < Hc; i += 256) { float d = x[i] - mean; vv += d * d; }
  red[t] = vv; __syncthreads();
  for (int o = 128; o > 0; o >>= 1) { if (t < o) red[t] += red[t + o]; __syncthreads(); }
  const float rstd = rsqrtf(red[0] * (1.0f / Hc) + EPSc);
  __syncthreads();
  for (int i = 2 * t; i < Hc; i += 512) {
    const float y0 = (x[i] - mean) * rstd * sN[i] + bN[i];
    const float y1 = (x[i + 1] - mean) * rstd * sN[i + 1] + bN[i + 1];
    Y[(size_t)row * Hc + i] = y0;
    Y[(size_t)row * Hc + i + 1] = y1;
    if (Ybf) *(unsigned int*)&Ybf[(size_t)row * Hc + i] = f2bf_pk(y0, y1);
  }
}

// ---------------------------------------------------------------------------
// Embedding gather + LayerNorm, dual f32 + bf16 outputs.
// ---------------------------------------------------------------------------
__global__ __launch_bounds__(256) void embed_ln(
    const float* __restrict__ we, const float* __restrict__ pe,
    const float* __restrict__ te, const float* __restrict__ lns,
    const float* __restrict__ lnb, const int* __restrict__ ids,
    const int* __restrict__ tts, float* __restrict__ out,
    unsigned short* __restrict__ outbf)
{
  __shared__ float xs[Hc];
  __shared__ float red[256];
  const int row = blockIdx.x, t = threadIdx.x;
  const int spos = row % Sc;
  const int id = ids[row], tt = tts[row];
  float s = 0.f;
  for (int i = t; i < Hc; i += 256) {
    float xv = we[(size_t)id * Hc + i] + pe[(size_t)spos * Hc + i] + te[(size_t)tt * Hc + i];
    xs[i] = xv; s += xv;
  }
  red[t] = s; __syncthreads();
  for (int o = 128; o > 0; o >>= 1) { if (t < o) red[t] += red[t + o]; __syncthreads(); }
  const float mean = red[0] * (1.0f / Hc);
  __syncthreads();
  float vv = 0.f;
  for (int i = t; i < Hc; i += 256) { float d = xs[i] - mean; vv += d * d; }
  red[t] = vv; __syncthreads();
  for (int o = 128; o > 0; o >>= 1) { if (t < o) red[t] += red[t + o]; __syncthreads(); }
  const float rstd = rsqrtf(red[0] * (1.0f / Hc) + EPSc);
  __syncthreads();
  for (int i = 2 * t; i < Hc; i += 512) {
    const float y0 = (xs[i] - mean) * rstd * lns[i] + lnb[i];
    const float y1 = (xs[i + 1] - mean) * rstd * lns[i + 1] + lnb[i + 1];
    out[(size_t)row * Hc + i] = y0;
    out[(size_t)row * Hc + i + 1] = y1;
    *(unsigned int*)&outbf[(size_t)row * Hc + i] = f2bf_pk(y0, y1);
  }
}

// ---------------------------------------------------------------------------
// Gate: bias1[row] = bias0[row] + GATE_K * sigmoid(LN(h[row]) . gate_w + gate_b)
// ---------------------------------------------------------------------------
__global__ __launch_bounds__(256) void gate_bias_k(
    const float* __restrict__ Hin, const float* __restrict__ lns,
    const float* __restrict__ lnb, const float* __restrict__ gw,
    const float* __restrict__ gb, const float* __restrict__ bias0,
    float* __restrict__ bias1)
{
  __shared__ float red[256];
  const int row = blockIdx.x, t = threadIdx.x;
  const float* x = Hin + (size_t)row * Hc;
  float s = 0.f;
  for (int i = t; i < Hc; i += 256) s += x[i];
  red[t] = s; __syncthreads();
  for (int o = 128; o > 0; o >>= 1) { if (t < o) red[t] += red[t + o]; __syncthreads(); }
  const float mean = red[0] * (1.0f / Hc);
  __syncthreads();
  float vv = 0.f;
  for (int i = t; i < Hc; i += 256) { float d = x[i] - mean; vv += d * d; }
  red[t] = vv; __syncthreads();
  for (int o = 128; o > 0; o >>= 1) { if (t < o) red[t] += red[t + o]; __syncthreads(); }
  const float rstd = rsqrtf(red[0] * (1.0f / Hc) + EPSc);
  __syncthreads();
  float dot = 0.f;
  for (int i = t; i < Hc; i += 256)
    dot += ((x[i] - mean) * rstd * lns[i] + lnb[i]) * gw[i];
  red[t] = dot; __syncthreads();
  for (int o = 128; o > 0; o >>= 1) { if (t < o) red[t] += red[t + o]; __syncthreads(); }
  if (t == 0) {
    const float z = red[0] + gb[0];
    const float gate = GATE_Kc / (1.0f + __expf(-z));
    bias1[row] = bias0[row] + gate;
  }
}

__global__ void mask_bias_k(const int* __restrict__ mask, float* __restrict__ bias0, int n)
{
  const int i = blockIdx.x * 256 + threadIdx.x;
  if (i < n) bias0[i] = (1.0f - (float)mask[i]) * -10000.0f;
}

// ---------------------------------------------------------------------------
extern "C" void kernel_launch(void* const* d_in, const int* in_sizes, int n_in,
                              void* d_out, int out_size, void* d_ws, size_t ws_size,
                              hipStream_t stream)
{
  (void)in_sizes; (void)n_in; (void)out_size; (void)ws_size;

  const float* word_emb  = (const float*)d_in[0];
  const float* pos_emb   = (const float*)d_in[1];
  const float* type_emb  = (const float*)d_in[2];
  const float* emb_ln_s  = (const float*)d_in[3];
  const float* emb_ln_b  = (const float*)d_in[4];
  const float* wq        = (const float*)d_in[5];
  const float* bq        = (const float*)d_in[6];
  const float* wk        = (const float*)d_in[7];
  const float* bk        = (const float*)d_in[8];
  const float* wv        = (const float*)d_in[9];
  const float* bv        = (const float*)d_in[10];
  const float* wo        = (const float*)d_in[11];
  const float* bo        = (const float*)d_in[12];
  const float* ln1_s     = (const float*)d_in[13];
  const float* ln1_b     = (const float*)d_in[14];
  const float* wi        = (const float*)d_in[15];
  const float* bi        = (const float*)d_in[16];
  const float* wo2       = (const float*)d_in[17];
  const float* bo2       = (const float*)d_in[18];
  const float* ln2_s     = (const float*)d_in[19];
  const float* ln2_b     = (const float*)d_in[20];
  const float* gate_ln_s = (const float*)d_in[21];
  const float* gate_ln_b = (const float*)d_in[22];
  const float* gate_w    = (const float*)d_in[23];
  const float* gate_b    = (const float*)d_in[24];
  const int* input_ids   = (const int*)d_in[25];
  const int* attn_mask   = (const int*)d_in[26];
  const int* token_type  = (const int*)d_in[27];
  float* out = (float*)d_out;

  // ---- workspace carve (all 16B aligned)
  const size_t MH = (size_t)Mc * Hc;          // 1,572,864
  const size_t MF = (size_t)Mc * FFc;         // 6,291,456
  char* p = (char*)d_ws;
  auto takeF = [&](size_t n) { float* r = (float*)p; p += n * 4; return r; };
  auto takeH = [&](size_t n) { unsigned short* r = (unsigned short*)p; p += n * 2; return r; };

  float* hbuf  = takeF(MH);   // hidden state f32 (residual)
  float* tmp   = takeF(MH);   // pre-LN residual sums
  float* af32  = takeF(MH);   // post-LN1 f32 (residual)
  float* bias0 = takeF(Mc);
  float* bias1 = takeF(Mc);
  unsigned short* hbf  = takeH(MH);  // bf16 activations
  unsigned short* abf  = takeH(MH);
  unsigned short* qbf  = takeH(MH);
  unsigned short* kbf  = takeH(MH);
  unsigned short* vbf  = takeH(MH);
  unsigned short* cxbf = takeH(MH);
  unsigned short* fbf  = takeH(MF);
  unsigned short* wtq  = takeH((size_t)Hc * Hc);   // per-layer transposed bf16 weights
  unsigned short* wtk  = takeH((size_t)Hc * Hc);
  unsigned short* wtv  = takeH((size_t)Hc * Hc);
  unsigned short* wto  = takeH((size_t)Hc * Hc);
  unsigned short* wti  = takeH((size_t)Hc * FFc);
  unsigned short* wto2 = takeH((size_t)FFc * Hc);

  mask_bias_k<<<(Mc + 255) / 256, 256, 0, stream>>>(attn_mask, bias0, Mc);
  embed_ln<<<Mc, 256, 0, stream>>>(word_emb, pos_emb, type_emb, emb_ln_s, emb_ln_b,
                                   input_ids, token_type, hbuf, hbf);

  const dim3 gH(Mc / GTM, Hc / GTN);    // 32 x 6
  const dim3 gF(Mc / GTM, FFc / GTN);   // 32 x 24
  const dim3 gAttn(Bc * NHc, Sc / 16);  // 48 x 32
  const dim3 gTHH(Hc / 32, Hc / 32);    // 24 x 24
  const dim3 gTHF(FFc / 32, Hc / 32);   // 96 x 24 (W[768,3072])
  const dim3 gTFH(Hc / 32, FFc / 32);   // 24 x 96 (W[3072,768])

  for (int i = 0; i < Lc; ++i) {
    const size_t wHH = (size_t)i * Hc * Hc;
    const size_t wHF = (size_t)i * Hc * FFc;

    convert_transpose_w<<<gTHH, 256, 0, stream>>>(wq + wHH, wtq, Hc, Hc);
    convert_transpose_w<<<gTHH, 256, 0, stream>>>(wk + wHH, wtk, Hc, Hc);
    convert_transpose_w<<<gTHH, 256, 0, stream>>>(wv + wHH, wtv, Hc, Hc);
    convert_transpose_w<<<gTHH, 256, 0, stream>>>(wo + wHH, wto, Hc, Hc);
    convert_transpose_w<<<gTHF, 256, 0, stream>>>(wi + wHF, wti, Hc, FFc);
    convert_transpose_w<<<gTFH, 256, 0, stream>>>(wo2 + wHF, wto2, FFc, Hc);

    gemm_bf16_wmma<0><<<gH, 256, 0, stream>>>(hbf, wtq, bq + i * Hc, nullptr, qbf, Hc, Hc);
    gemm_bf16_wmma<0><<<gH, 256, 0, stream>>>(hbf, wtk, bk + i * Hc, nullptr, kbf, Hc, Hc);
    gemm_bf16_wmma<0><<<gH, 256, 0, stream>>>(hbf, wtv, bv + i * Hc, nullptr, vbf, Hc, Hc);

    const int sm1 = (i > GATE_LAYERc) ? 1 : 0;
    attention_wmma<<<gAttn, 32, 0, stream>>>(qbf, kbf, vbf, sm1 ? bias1 : bias0, cxbf, sm1);

    gemm_bf16_wmma<2><<<gH, 256, 0, stream>>>(cxbf, wto, bo + i * Hc, hbuf, tmp, Hc, Hc);
    layernorm_rows<<<Mc, 256, 0, stream>>>(tmp, ln1_s + i * Hc, ln1_b + i * Hc, af32, abf);

    gemm_bf16_wmma<1><<<gF, 256, 0, stream>>>(abf, wti, bi + i * FFc, nullptr, fbf, FFc, Hc);
    gemm_bf16_wmma<2><<<gH, 256, 0, stream>>>(fbf, wto2, bo2 + i * Hc, af32, tmp, Hc, FFc);

    float* dst = (i == Lc - 1) ? out : hbuf;
    unsigned short* dstbf = (i == Lc - 1) ? nullptr : hbf;
    layernorm_rows<<<Mc, 256, 0, stream>>>(tmp, ln2_s + i * Hc, ln2_b + i * Hc, dst, dstbf);

    if (i == GATE_LAYERc)
      gate_bias_k<<<Mc, 256, 0, stream>>>(hbuf, gate_ln_s, gate_ln_b, gate_w, gate_b,
                                          bias0, bias1);
  }
}
